// SelfAttention_56470230008429
// MI455X (gfx1250) — matrix operand order repaired and checked by reading, not compiled, not run on hardware
//
#include <hip/hip_runtime.h>

#define BB 4
#define SS 2048
#define DD 1024

typedef __bf16 bf16_t;
typedef __attribute__((ext_vector_type(16))) __bf16 v16bf;
typedef __attribute__((ext_vector_type(8)))  __bf16 v8bf;
typedef __attribute__((ext_vector_type(4)))  __bf16 v4bf;
typedef __attribute__((ext_vector_type(8)))  float  v8f;
typedef __attribute__((ext_vector_type(4)))  float  v4f;
typedef __attribute__((ext_vector_type(4)))  unsigned int v4u;
typedef __attribute__((ext_vector_type(8)))  int    v8i;
typedef __attribute__((ext_vector_type(4)))  int    v4i;

union V16BF { v16bf v; v8bf h[2]; };

static __device__ __forceinline__ v8f wmma_bf16(v16bf a, v16bf b, v8f c) {
  // D = A(16x32) * B(32x16) + C, fp32 accumulate
  return __builtin_amdgcn_wmma_f32_16x16x32_bf16(false, a, false, b,
                                                 (short)0, c, false, false);
}

// A-operand: lane (lm,lh) = row lm, K run {k+lh*16 .. +15} (contiguous)
static __device__ __forceinline__ v16bf loadA_f32_to_bf16(const float* row,
                                                          int k, int lh) {
  v4f a0 = *(const v4f*)(row + k + lh * 16);
  v4f a1 = *(const v4f*)(row + k + lh * 16 + 4);
  v4f a2 = *(const v4f*)(row + k + lh * 16 + 8);
  v4f a3 = *(const v4f*)(row + k + lh * 16 + 12);
  v16bf A;
#pragma unroll
  for (int i = 0; i < 4; ++i) {
    A[i]      = (bf16_t)a0[i];
    A[i + 4]  = (bf16_t)a1[i];
    A[i + 8]  = (bf16_t)a2[i];
    A[i + 12] = (bf16_t)a3[i];
  }
  return A;
}

// B-operand: lane (lm,lh) = col lm, K run {k+lh*16 .. +15} (contiguous)
static __device__ __forceinline__ v16bf loadB_f32_to_bf16(const float* row,
                                                          int k, int lh) {
  v4f b0 = *(const v4f*)(row + k + lh * 16);
  v4f b1 = *(const v4f*)(row + k + lh * 16 + 4);
  v4f b2 = *(const v4f*)(row + k + lh * 16 + 8);
  v4f b3 = *(const v4f*)(row + k + lh * 16 + 12);
  v16bf B;
#pragma unroll
  for (int i = 0; i < 4; ++i) {
    B[i]      = (bf16_t)b0[i];
    B[i + 4]  = (bf16_t)b1[i];
    B[i + 8]  = (bf16_t)b2[i];
    B[i + 12] = (bf16_t)b3[i];
  }
  return B;
}

// C layout: lane (lm,lh) holds col n0+lm, rows m0 + lh*8 + r (r=0..7)
static __device__ __forceinline__ void store_q_tile(bf16_t* q, int m0, int n0,
                                                    int lh, int lm, v8f acc) {
  bf16_t* qp = q + (size_t)(m0 + lh * 8) * DD + n0 + lm;
#pragma unroll
  for (int r = 0; r < 8; ++r)
    qp[(size_t)r * DD] = (bf16_t)(acc[r] * 0.03125f);    // 1/sqrt(D)
}

static __device__ __forceinline__ void store_kv_tile(bf16_t* krT, bf16_t* kvT,
                                                     int m0, int n0, int lh,
                                                     int lm, v8f acc) {
  const int b  = m0 >> 11;                // 2048 rows per batch
  const int sb = (m0 & 2047) + lh * 8;    // s of r=0 (even)
  v8bf pk;
#pragma unroll
  for (int r = 0; r < 8; ++r) pk[r] = (bf16_t)acc[r];
  // kvT[b][d=n0+lm][s=sb..sb+7] : one contiguous b128 store
  *(v8bf*)(kvT + ((size_t)b * DD + n0 + lm) * SS + sb) = pk;
  // krT[b][t][d]: flat = s*1024 + col -> t = (s&1)*1024 + col, d = s>>1
  v4bf ev, ov;
#pragma unroll
  for (int i = 0; i < 4; ++i) { ev[i] = pk[2 * i]; ov[i] = pk[2 * i + 1]; }
  const size_t db = (size_t)(sb >> 1);
  *(v4bf*)(krT + ((size_t)b * SS + (n0 + lm)) * DD + db)        = ev;
  *(v4bf*)(krT + ((size_t)b * SS + 1024 + (n0 + lm)) * DD + db) = ov;
}

// ---------------------------------------------------------------------------
// Kernel 1: q = (x @ wq^T)/32 -> bf16 [B*S, D];  kv = x @ wv^T -> krT, kvT.
// 32x32 register block per wave (4 C tiles): 4 WMMA per 16 b128 loads.
// ---------------------------------------------------------------------------
__global__ __launch_bounds__(256) void qkv_gemm_kernel(
    const float* __restrict__ x, const float* __restrict__ wq,
    const float* __restrict__ wv, bf16_t* __restrict__ q,
    bf16_t* __restrict__ krT, bf16_t* __restrict__ kvT) {
  const int lane = threadIdx.x & 31;
  const int wave = threadIdx.x >> 5;
  int tile = blockIdx.x * 8 + wave;           // 16384 tiles total
  const int n0 = (tile & 31) * 32;  tile >>= 5;   // 32 col blocks
  const int m0 = (tile & 255) * 32; tile >>= 8;   // 256 row blocks
  const int isKV = tile & 1;
  const int lh = lane >> 4;
  const int lm = lane & 15;
  const float* w = isKV ? wv : wq;

  const float* arow0 = x + (size_t)(m0 + lm) * DD;
  const float* arow1 = arow0 + (size_t)16 * DD;
  const float* brow0 = w + (size_t)(n0 + lm) * DD;
  const float* brow1 = brow0 + (size_t)16 * DD;

  v8f c00 = {}, c01 = {}, c10 = {}, c11 = {};
  for (int k = 0; k < DD; k += 32) {
    __builtin_prefetch(arow0 + k + 256, 0, 1);   // global_prefetch_b8
    v16bf A0 = loadA_f32_to_bf16(arow0, k, lh);
    v16bf A1 = loadA_f32_to_bf16(arow1, k, lh);
    v16bf B0 = loadB_f32_to_bf16(brow0, k, lh);
    v16bf B1 = loadB_f32_to_bf16(brow1, k, lh);
    c00 = wmma_bf16(A0, B0, c00);
    c01 = wmma_bf16(A0, B1, c01);
    c10 = wmma_bf16(A1, B0, c10);
    c11 = wmma_bf16(A1, B1, c11);
  }

  if (!isKV) {
    store_q_tile(q, m0,      n0,      lh, lm, c00);
    store_q_tile(q, m0,      n0 + 16, lh, lm, c01);
    store_q_tile(q, m0 + 16, n0,      lh, lm, c10);
    store_q_tile(q, m0 + 16, n0 + 16, lh, lm, c11);
  } else {
    store_kv_tile(krT, kvT, m0,      n0,      lh, lm, c00);
    store_kv_tile(krT, kvT, m0,      n0 + 16, lh, lm, c01);
    store_kv_tile(krT, kvT, m0 + 16, n0,      lh, lm, c10);
    store_kv_tile(krT, kvT, m0 + 16, n0 + 16, lh, lm, c11);
  }
}

// ---------------------------------------------------------------------------
// Kernel 2: per block: batch b, 16 query rows, 8 waves.
//  q block (16x1024 bf16, contiguous 32KB) staged to LDS via TDM.
//  phase 1: wave computes 256-wide t slice of scores (k-outer: A reused 16x)
//  phase 2: full-row softmax stats over ALL t; post-softmax tril -> bf16 p
//  phase 3: wave computes 128-wide d slice of p @ kv (k-outer: A reused 8x)
// ---------------------------------------------------------------------------
__global__ __launch_bounds__(256) void attn_kernel(
    const bf16_t* __restrict__ q, const bf16_t* __restrict__ krT,
    const bf16_t* __restrict__ kvT, float* __restrict__ out) {
  __shared__ bf16_t qs[16 * DD];   // 32 KB
  __shared__ bf16_t ps[16 * SS];   // 64 KB
  __shared__ float  red[8 * 16];
  __shared__ float  gmax[16];
  __shared__ float  gsum[16];

  const int b    = blockIdx.x >> 7;
  const int s0   = (blockIdx.x & 127) * 16;
  const int lane = threadIdx.x & 31;
  const int wave = threadIdx.x >> 5;
  const int lh   = lane >> 4;
  const int lm   = lane & 15;

  const bf16_t* qg = q + ((size_t)b * SS + s0) * DD;

#if __has_builtin(__builtin_amdgcn_tensor_load_to_lds)
  // TDM: one contiguous 32KB tile, D# per CDNA5 ISA 8.3-8.4.
  if (threadIdx.x < 32) {     // wave 0 issues; TDM ignores EXEC
    unsigned long long ga = (unsigned long long)(const void*)qg;
    unsigned int lds_off  = (unsigned int)(unsigned long long)(const void*)qs;
    v4u g0 = { 1u,                                     // count=1, user D#
               lds_off,                                // lds_addr
               (unsigned int)ga,                       // global_addr[31:0]
               (unsigned int)((ga >> 32) & 0x01FFFFFFull) | (2u << 30) };
    v8i g1 = { 0x00010000,   // workgroup_mask=0, data_size=1 (2B)
               0x40000000,   // tensor_dim0[15:0]=0x4000 in bits[63:48]
               0x00010000,   // tensor_dim0 hi=0, tensor_dim1=1
               0x40000000,   // tile_dim0 = 0x4000 (16384 elements)
               1,            // tile_dim1 = 1
               16384,        // tensor_dim0_stride lo
               0, 0 };
    v4i gz = { 0, 0, 0, 0 };
#if defined(__clang_major__) && __clang_major__ >= 23
    v8i gz8 = { 0, 0, 0, 0, 0, 0, 0, 0 };
    __builtin_amdgcn_tensor_load_to_lds(g0, g1, gz, gz, gz8, 0);
#else
    __builtin_amdgcn_tensor_load_to_lds(g0, g1, gz, gz, 0);
#endif
    __builtin_amdgcn_s_wait_tensorcnt(0);
  }
#else
  for (int i = threadIdx.x; i < 16 * DD / 8; i += 256)
    ((v8bf*)qs)[i] = ((const v8bf*)qg)[i];
#endif
  __syncthreads();

  // ---- phase 1: scores (already scaled by 1/32 via q) ----
  const int t0 = wave * 256;
  const bf16_t* krb = krT + (size_t)b * SS * DD;
  const bf16_t* bbase = krb + (size_t)(t0 + lm) * DD + lh * 16;
  v8f sc[16];
  {
    v8f z = {};
#pragma unroll
    for (int j = 0; j < 16; ++j) sc[j] = z;
  }
  for (int k = 0; k < DD; k += 32) {
    V16BF A;
    A.h[0] = *(const v8bf*)(qs + lm * DD + k + lh * 16);
    A.h[1] = *(const v8bf*)(qs + lm * DD + k + lh * 16 + 8);
#pragma unroll
    for (int j = 0; j < 16; ++j) {
      v16bf Bv = *(const v16bf*)(bbase + (size_t)j * (16 * DD) + k);
      sc[j] = wmma_bf16(A.v, Bv, sc[j]);
    }
  }

  // ---- phase 2: row max over all t ----
  float pm[8];
#pragma unroll
  for (int r = 0; r < 8; ++r) {
    float m = sc[0][r];
#pragma unroll
    for (int j = 1; j < 16; ++j) m = fmaxf(m, sc[j][r]);
    pm[r] = m;
  }
#pragma unroll
  for (int d = 1; d <= 8; d <<= 1)
#pragma unroll
    for (int r = 0; r < 8; ++r) pm[r] = fmaxf(pm[r], __shfl_xor(pm[r], d, 32));
  if (lm == 0) {
#pragma unroll
    for (int r = 0; r < 8; ++r) red[wave * 16 + lh * 8 + r] = pm[r];
  }
  __syncthreads();
  if (threadIdx.x < 16) {
    float m = red[threadIdx.x];
    for (int wv = 1; wv < 8; ++wv) m = fmaxf(m, red[wv * 16 + threadIdx.x]);
    gmax[threadIdx.x] = m;
  }
  __syncthreads();

  float rmax[8], psum[8];
#pragma unroll
  for (int r = 0; r < 8; ++r) { rmax[r] = gmax[r + 8 * lh]; psum[r] = 0.f; }

  // exp; denominator over ALL t (ref semantics); tril applied post-softmax
#pragma unroll
  for (int j = 0; j < 16; ++j) {
    const int t = t0 + j * 16 + lm;
#pragma unroll
    for (int r = 0; r < 8; ++r) {
      const int row = r + 8 * lh;
      float e = __expf(sc[j][r] - rmax[r]);
      psum[r] += e;
      ps[row * SS + t] = (t <= s0 + row) ? (bf16_t)e : (bf16_t)0.f;
    }
  }
#pragma unroll
  for (int d = 1; d <= 8; d <<= 1)
#pragma unroll
    for (int r = 0; r < 8; ++r) psum[r] += __shfl_xor(psum[r], d, 32);
  if (lm == 0) {
#pragma unroll
    for (int r = 0; r < 8; ++r) red[wave * 16 + lh * 8 + r] = psum[r];
  }
  __syncthreads();
  if (threadIdx.x < 16) {
    float s = red[threadIdx.x];
    for (int wv = 1; wv < 8; ++wv) s += red[wv * 16 + threadIdx.x];
    gsum[threadIdx.x] = s;
  }
  __syncthreads();

  // ---- phase 3: out = (p @ kv) / row_sum ----
  float rs[8];
#pragma unroll
  for (int r = 0; r < 8; ++r) rs[r] = 1.0f / gsum[r + 8 * lh];

  const bf16_t* kvb = kvT + (size_t)b * DD * SS;
  const int d0 = wave * 128;
  const bf16_t* vbase = kvb + (size_t)(d0 + lm) * SS + lh * 16;
  float* outb = out + ((size_t)b * SS + s0) * DD;

  v8f oa[8];
  {
    v8f z = {};
#pragma unroll
    for (int jn = 0; jn < 8; ++jn) oa[jn] = z;
  }
  for (int k = 0; k < SS; k += 32) {
    V16BF A;
    A.h[0] = *(const v8bf*)(ps + lm * SS + k + lh * 16);
    A.h[1] = *(const v8bf*)(ps + lm * SS + k + lh * 16 + 8);
#pragma unroll
    for (int jn = 0; jn < 8; ++jn) {
      v16bf Bv = *(const v16bf*)(vbase + (size_t)jn * (16 * SS) + k);
      oa[jn] = wmma_bf16(A.v, Bv, oa[jn]);
    }
  }
#pragma unroll
  for (int jn = 0; jn < 8; ++jn) {
    const int dc = d0 + jn * 16 + lm;
#pragma unroll
    for (int r = 0; r < 8; ++r)
      outb[(size_t)(r + 8 * lh) * DD + dc] = oa[jn][r] * rs[r];
  }
}

// ---------------------------------------------------------------------------
extern "C" void kernel_launch(void* const* d_in, const int* in_sizes, int n_in,
                              void* d_out, int out_size, void* d_ws, size_t ws_size,
                              hipStream_t stream) {
  (void)in_sizes; (void)n_in; (void)out_size; (void)ws_size;
  const float* x  = (const float*)d_in[0];
  const float* wq = (const float*)d_in[1];
  const float* wv = (const float*)d_in[2];

  // ws layout: q bf16 (16MB) | krT bf16 (16MB) | kvT bf16 (16MB) = 48 MB
  bf16_t* qb  = (bf16_t*)d_ws;
  bf16_t* krT = qb  + (size_t)BB * SS * DD;
  bf16_t* kvT = krT + (size_t)BB * SS * DD;

  // 2 outputs * 256 row-blocks * 32 col-blocks = 16384 wave-tiles / 8 waves
  qkv_gemm_kernel<<<2048, 256, 0, stream>>>(x, wq, wv, qb, krT, kvT);
  attn_kernel<<<BB * (SS / 16), 256, 0, stream>>>(qb, krT, kvT, (float*)d_out);
}
